// ManualCausalSelfAttention_40149354283752
// MI455X (gfx1250) — matrix operand order repaired and checked by reading, not compiled, not run on hardware
//
#include <hip/hip_runtime.h>
#include <stdint.h>

#define HIDDEN 1024
#define HEADS  16
#define HDIM   64
#define SEQ    2048
#define BATCH  4
#define ROWS   (BATCH*SEQ)      // 8192
#define NQKV   (3*HIDDEN)       // 3072

typedef __attribute__((ext_vector_type(16))) __bf16 v16bf;
typedef __attribute__((ext_vector_type(8)))  float  v8f;

union AB { v16bf v; uint32_t u[8]; };

// ---- bf16 helpers (bit-level, no ABI dependence) ----
__device__ __forceinline__ float bf2f(uint16_t h) {
    union { uint32_t u; float f; } c; c.u = ((uint32_t)h) << 16; return c.f;
}
__device__ __forceinline__ uint16_t f2bf(float f) {
    union { float f; uint32_t u; } c; c.f = f;
    uint32_t u = c.u + 0x7FFFu + ((c.u >> 16) & 1u);   // round-to-nearest-even
    return (uint16_t)(u >> 16);
}

// ---- WMMA operand index helpers (16-bit ops, wave32; this code's contiguous k = 16*(lane/16) + 2j for A and B) ----
__device__ __forceinline__ int a_k0(int lane, int j) {
    int half = lane >> 4;
    return half * 16 + 2 * j;
}
__device__ __forceinline__ int b_k0(int lane, int j) {
    return ((lane >> 4) << 4) + 2 * j;
}

__device__ __forceinline__ v8f wmma_bf16(v16bf a, v16bf b, v8f c) {
    return __builtin_amdgcn_wmma_f32_16x16x32_bf16(false, a, false, b, (short)0, c, false, false);
}

__device__ __forceinline__ v8f vzero8() {
    v8f z;
    #pragma unroll
    for (int e = 0; e < 8; ++e) z[e] = 0.0f;
    return z;
}

// ---- CDNA5 async global -> LDS (ASYNCcnt-tracked), ISA 10 / 15.18.3 ----
__device__ __forceinline__ void async_load_b128(uint32_t lds_off, const void* gaddr) {
    asm volatile("global_load_async_to_lds_b128 %0, %1, off"
                 :: "v"(lds_off), "v"(gaddr) : "memory");
}
__device__ __forceinline__ void wait_async0() {
    asm volatile("s_wait_asynccnt 0" ::: "memory");
}

// ================= conversion / transpose =================
__global__ void k_f32_to_bf16(const float* __restrict__ in, uint16_t* __restrict__ out, int n) {
    int i = blockIdx.x * blockDim.x + threadIdx.x;
    if (i < n) out[i] = f2bf(in[i]);
}

// in [K,N] f32 -> out [N,K] bf16
__global__ void k_transpose_bf16(const float* __restrict__ in, uint16_t* __restrict__ out, int K, int N) {
    int i = blockIdx.x * blockDim.x + threadIdx.x;
    if (i < K * N) {
        int k = i / N, n = i % N;
        out[(size_t)n * K + k] = f2bf(in[i]);
    }
}

// =========================================================================
// Shared 128x64 GEMM core: 8 waves, each wave one 16-row strip, all waves
// share the 64-col weight tile staged in double-buffered LDS via async DMA.
// A in registers (double-buffered), B from LDS in B-operand layout.
// =========================================================================
__device__ __forceinline__ void load_a16x32(AB& a, const uint16_t* __restrict__ A,
                                            int arow, int K, int kk, int lane) {
    #pragma unroll
    for (int j = 0; j < 8; ++j) {
        int k0 = a_k0(lane, j);
        a.u[j] = *(const uint32_t*)(A + (size_t)arow * K + kk + k0);
    }
}

__device__ __forceinline__ void stage_btile(const uint16_t* __restrict__ BT, int col0, int K,
                                            int kk, uint16_t* buf) {
    // 64 rows (cols of C) x 32 bf16 (64B) = 4KB; 256 threads x 16B async each
    int tid = threadIdx.x;
    int row = tid >> 2, chunk = tid & 3;
    uint32_t lds_off = (uint32_t)(uintptr_t)buf + (uint32_t)(row * 64 + chunk * 16);
    const uint16_t* g = BT + (size_t)(col0 + row) * K + kk + chunk * 8;
    async_load_b128(lds_off, g);
}

__device__ __forceinline__ void consume_btile(const uint16_t* buf, AB a, v8f (&acc)[4],
                                              int lane) {
    const int ln = lane & 15, half = lane >> 4;
    AB b[4];
    #pragma unroll
    for (int tt = 0; tt < 4; ++tt) {
        const uint16_t* base = buf + (size_t)(tt * 16 + ln) * 32 + half * 16;
        #pragma unroll
        for (int j = 0; j < 8; ++j)
            b[tt].u[j] = *(const uint32_t*)(base + 2 * j);
    }
    #pragma unroll
    for (int tt = 0; tt < 4; ++tt)
        acc[tt] = wmma_bf16(a.v, b[tt].v, acc[tt]);
}

__device__ __forceinline__ void gemm_128x64_core(const uint16_t* __restrict__ A,
                                                 const uint16_t* __restrict__ BT,
                                                 int row0_wave, int col0, int K,
                                                 uint16_t (&bstage)[2][64 * 32],
                                                 v8f (&acc)[4]) {
    const int lane = threadIdx.x & 31;
    const int arow = row0_wave + (lane & 15);

    stage_btile(BT, col0, K, 0, bstage[0]);
    wait_async0();
    __syncthreads();

    AB a_cur; load_a16x32(a_cur, A, arow, K, 0, lane);
    AB a_nxt;

    // K % 64 == 0 for all our shapes
    for (int kk = 0; kk < K; kk += 64) {
        // ---- sub-step 0: consume bstage[0], stage bstage[1] for kk+32 ----
        stage_btile(BT, col0, K, kk + 32, bstage[1]);
        load_a16x32(a_nxt, A, arow, K, kk + 32, lane);
        consume_btile(bstage[0], a_cur, acc, lane);
        a_cur = a_nxt;
        wait_async0();
        __syncthreads();

        // ---- sub-step 1: consume bstage[1], stage bstage[0] for kk+64 ----
        if (kk + 64 < K) {
            stage_btile(BT, col0, K, kk + 64, bstage[0]);
            load_a16x32(a_nxt, A, arow, K, kk + 64, lane);
        } else {
            a_nxt = a_cur;
        }
        consume_btile(bstage[1], a_cur, acc, lane);
        a_cur = a_nxt;
        wait_async0();
        __syncthreads();
    }
}

// ================= QKV GEMM: [8192,1024] x [1024,3072] -> qh/kh (B,H,S,D), vt (B,H,D,S) =================
__global__ void __launch_bounds__(256)
k_gemm_qkv(const uint16_t* __restrict__ A,    // x bf16 [ROWS, HIDDEN]
           const uint16_t* __restrict__ BT,   // Wqkv^T bf16 [NQKV, HIDDEN]
           const float*    __restrict__ bias, // [NQKV]
           uint16_t* __restrict__ qh, uint16_t* __restrict__ kh, uint16_t* __restrict__ vt) {
    __shared__ uint16_t bstage[2][64 * 32];
    const int lane = threadIdx.x & 31;
    const int wave = threadIdx.x >> 5;
    const int MB   = ROWS / 128;                // 64 row blocks
    const int mb   = blockIdx.x % MB;
    const int nt   = blockIdx.x / MB;           // [0,48)
    const int row0 = mb * 128 + wave * 16;
    const int col0 = nt * 64;
    const int half = lane >> 4, ln = lane & 15;

    v8f acc[4] = { vzero8(), vzero8(), vzero8(), vzero8() };
    gemm_128x64_core(A, BT, row0, col0, HIDDEN, bstage, acc);

    // scatter into head-major layouts
    #pragma unroll
    for (int tt = 0; tt < 4; ++tt) {
        int ng = col0 + tt * 16 + ln;
        float bs = bias[ng];
        int which = ng >> 10;            // 0=q 1=k 2=v
        int rem   = ng & 1023;
        int h = rem >> 6, d = rem & 63;
        #pragma unroll
        for (int r = 0; r < 8; ++r) {
            int rg = row0 + r + 8 * half;
            int b_ = rg >> 11, s = rg & 2047;
            uint16_t val = f2bf(acc[tt][r] + bs);
            size_t idx_h = (((size_t)(b_ * HEADS + h) * SEQ) + s) * HDIM + d;
            if (which == 0)      qh[idx_h] = val;
            else if (which == 1) kh[idx_h] = val;
            else vt[(((size_t)(b_ * HEADS + h) * HDIM) + d) * SEQ + s] = val;
        }
    }
}

// ================= output projection: [8192,1024] x [1024,1024] + bias, padding, f32 out =================
__global__ void __launch_bounds__(256)
k_gemm_proj(const uint16_t* __restrict__ A,        // attn bf16 [ROWS, HIDDEN]
            const uint16_t* __restrict__ BT,       // Wproj^T bf16 [HIDDEN, HIDDEN]
            const float*    __restrict__ bias,     // [HIDDEN]
            const unsigned char* __restrict__ pad, // [ROWS] bool
            float* __restrict__ out) {
    __shared__ uint16_t bstage[2][64 * 32];
    const int lane = threadIdx.x & 31;
    const int wave = threadIdx.x >> 5;
    const int MB   = ROWS / 128;                // 64
    const int mb   = blockIdx.x % MB;
    const int nt   = blockIdx.x / MB;           // [0,16)
    const int row0 = mb * 128 + wave * 16;
    const int col0 = nt * 64;
    const int half = lane >> 4, ln = lane & 15;

    v8f acc[4] = { vzero8(), vzero8(), vzero8(), vzero8() };
    gemm_128x64_core(A, BT, row0, col0, HIDDEN, bstage, acc);

    #pragma unroll
    for (int tt = 0; tt < 4; ++tt) {
        int ng = col0 + tt * 16 + ln;
        float bs = bias[ng];
        #pragma unroll
        for (int r = 0; r < 8; ++r) {
            int rg = row0 + r + 8 * half;
            float pm = pad[rg] ? 1.0f : 0.0f;
            out[(size_t)rg * HIDDEN + ng] = (acc[tt][r] + bs) * pm;
        }
    }
}

// ================= RoPE (in place on qh, kh) =================
__global__ void k_rope(uint16_t* __restrict__ qh, uint16_t* __restrict__ kh) {
    int idx = blockIdx.x * blockDim.x + threadIdx.x;  // over B*H*S*32
    int i  = idx & 31;                                // dim pair
    int s  = (idx >> 5) & (SEQ - 1);
    int bh = idx >> 16;
    if (bh >= BATCH * HEADS) return;
    float inv = __powf(10000.0f, -(float)(2 * i) / 64.0f);
    float ang = (float)s * inv;
    float sn, cs; __sincosf(ang, &sn, &cs);
    size_t base = ((size_t)bh * SEQ + s) * HDIM + 2 * i;
    float qe = bf2f(qh[base]), qo = bf2f(qh[base + 1]);
    qh[base]     = f2bf(qe * cs - qo * sn);
    qh[base + 1] = f2bf(qo * cs + qe * sn);
    float ke = bf2f(kh[base]), ko = bf2f(kh[base + 1]);
    kh[base]     = f2bf(ke * cs - ko * sn);
    kh[base + 1] = f2bf(ko * cs + ke * sn);
}

// ================= flash attention: one wave = 16 q rows, 32-key blocks =================
__global__ void __launch_bounds__(256)
k_attn(const uint16_t* __restrict__ qh,
       const uint16_t* __restrict__ kh,
       const uint16_t* __restrict__ vt,
       uint16_t* __restrict__ out /* bf16 [ROWS, HIDDEN], heads merged */) {
    __shared__ uint16_t ldsP[8][16 * 32];     // per-wave P scratch (C-layout -> A-layout bridge)
    const int lane  = threadIdx.x & 31;
    const int wave  = threadIdx.x >> 5;
    const int w     = blockIdx.x * 8 + wave;
    const int QT    = SEQ / 16;               // 128
    const int qtile = w % QT;
    const int bh    = w / QT;                 // 0..63
    const int q0    = qtile * 16;
    const int half  = lane >> 4, ln = lane & 15;

    // Q A-operands: 16x64 split into K=0..31 and K=32..63
    AB qa0, qa1;
    #pragma unroll
    for (int j = 0; j < 8; ++j) {
        int k0 = a_k0(lane, j);
        size_t rowbase = ((size_t)bh * SEQ + q0 + ln) * HDIM;
        qa0.u[j] = *(const uint32_t*)(qh + rowbase + k0);
        qa1.u[j] = *(const uint32_t*)(qh + rowbase + 32 + k0);
    }

    v8f o[4] = { vzero8(), vzero8(), vzero8(), vzero8() };
    float mi[8], li[8];
    #pragma unroll
    for (int r = 0; r < 8; ++r) { mi[r] = -3.0e38f; li[r] = 0.0f; }

    const int nkb = (q0 + 15) / 32 + 1;       // causal: key blocks of 32
    for (int kbi = 0; kbi < nkb; ++kbi) {
        const int key0 = kbi * 32;

        // ---- S = Q * K^T (16x32), two 16x16 C tiles ----
        AB kb[4];
        #pragma unroll
        for (int cg = 0; cg < 2; ++cg) {
            size_t rowbase = ((size_t)bh * SEQ + key0 + cg * 16 + ln) * HDIM;
            #pragma unroll
            for (int j = 0; j < 8; ++j) {
                int k0 = b_k0(lane, j);
                kb[cg * 2 + 0].u[j] = *(const uint32_t*)(kh + rowbase + k0);
                kb[cg * 2 + 1].u[j] = *(const uint32_t*)(kh + rowbase + 32 + k0);
            }
        }
        v8f s0 = wmma_bf16(qa0.v, kb[0].v, wmma_bf16(qa1.v, kb[1].v, vzero8()));
        v8f s1 = wmma_bf16(qa0.v, kb[2].v, wmma_bf16(qa1.v, kb[3].v, vzero8()));

        // ---- scale + causal mask + online softmax ----
        float nm[8];
        #pragma unroll
        for (int r = 0; r < 8; ++r) {
            int qrow = q0 + r + 8 * half;
            int keyA = key0 + ln, keyB = key0 + 16 + ln;
            float a  = (keyA <= qrow) ? s0[r] * 0.125f : -3.0e38f;
            float b_ = (keyB <= qrow) ? s1[r] * 0.125f : -3.0e38f;
            s0[r] = a; s1[r] = b_;
            float mx = fmaxf(a, b_);
            mx = fmaxf(mx, __shfl_xor(mx, 1, 32));
            mx = fmaxf(mx, __shfl_xor(mx, 2, 32));
            mx = fmaxf(mx, __shfl_xor(mx, 4, 32));
            mx = fmaxf(mx, __shfl_xor(mx, 8, 32));
            nm[r] = fmaxf(mi[r], mx);
        }
        float alpha[8];
        #pragma unroll
        for (int r = 0; r < 8; ++r) {
            alpha[r] = __expf(mi[r] - nm[r]);
            float p0 = __expf(s0[r] - nm[r]);
            float p1 = __expf(s1[r] - nm[r]);
            s0[r] = p0; s1[r] = p1;
            float sum = p0 + p1;
            sum += __shfl_xor(sum, 1, 32);
            sum += __shfl_xor(sum, 2, 32);
            sum += __shfl_xor(sum, 4, 32);
            sum += __shfl_xor(sum, 8, 32);
            li[r] = li[r] * alpha[r] + sum;
            mi[r] = nm[r];
        }
        #pragma unroll
        for (int t = 0; t < 4; ++t)
            #pragma unroll
            for (int r = 0; r < 8; ++r) o[t][r] *= alpha[r];

        // ---- C-layout P -> LDS -> A-layout bf16 (per-wave region, in-order DS) ----
        #pragma unroll
        for (int r = 0; r < 8; ++r) {
            int row = r + 8 * half;
            ldsP[wave][row * 32 + ln]      = f2bf(s0[r]);
            ldsP[wave][row * 32 + 16 + ln] = f2bf(s1[r]);
        }
        asm volatile("s_wait_dscnt 0" ::: "memory");
        AB pa;
        #pragma unroll
        for (int j = 0; j < 8; ++j) {
            int k0 = a_k0(lane, j);
            pa.u[j] = *(const uint32_t*)(&ldsP[wave][ln * 32 + k0]);
        }

        // ---- O += P * V : batch all V operands, then 4 back-to-back WMMAs ----
        AB vb[4];
        #pragma unroll
        for (int t = 0; t < 4; ++t) {
            #pragma unroll
            for (int j = 0; j < 8; ++j) {
                int k0 = b_k0(lane, j);
                vb[t].u[j] = *(const uint32_t*)(vt + ((size_t)bh * HDIM + t * 16 + ln) * SEQ + key0 + k0);
            }
        }
        #pragma unroll
        for (int t = 0; t < 4; ++t)
            o[t] = wmma_bf16(pa.v, vb[t].v, o[t]);
    }

    // ---- normalize and merge heads: out[b, s, h*64 + d] ----
    int b_ = bh / HEADS, h = bh % HEADS;
    #pragma unroll
    for (int t = 0; t < 4; ++t) {
        #pragma unroll
        for (int r = 0; r < 8; ++r) {
            int row = q0 + r + 8 * half;
            float val = o[t][r] / li[r];
            out[((size_t)(b_ * SEQ + row)) * HIDDEN + h * HDIM + t * 16 + ln] = f2bf(val);
        }
    }
}

// ================= launcher =================
extern "C" void kernel_launch(void* const* d_in, const int* in_sizes, int n_in,
                              void* d_out, int out_size, void* d_ws, size_t ws_size,
                              hipStream_t stream) {
    const float* x            = (const float*)d_in[0];
    const unsigned char* pad  = (const unsigned char*)d_in[1];
    const float* Wqkv         = (const float*)d_in[2];
    const float* bqkv         = (const float*)d_in[3];
    const float* Wproj        = (const float*)d_in[4];
    const float* bproj        = (const float*)d_in[5];
    float* out                = (float*)d_out;

    char* ws = (char*)d_ws;
    size_t off = 0;
    auto alloc = [&](size_t bytes) -> void* {
        void* p = ws + off;
        off += (bytes + 255) & ~(size_t)255;
        return p;
    };
    uint16_t* xb     = (uint16_t*)alloc((size_t)ROWS * HIDDEN * 2);
    uint16_t* wqkvT  = (uint16_t*)alloc((size_t)NQKV * HIDDEN * 2);
    uint16_t* wprojT = (uint16_t*)alloc((size_t)HIDDEN * HIDDEN * 2);
    uint16_t* qh     = (uint16_t*)alloc((size_t)ROWS * HIDDEN * 2);
    uint16_t* kh     = (uint16_t*)alloc((size_t)ROWS * HIDDEN * 2);
    uint16_t* vtb    = (uint16_t*)alloc((size_t)ROWS * HIDDEN * 2);
    uint16_t* attn   = (uint16_t*)alloc((size_t)ROWS * HIDDEN * 2);

    // stage 0: precision conversion + weight transposes
    k_f32_to_bf16<<<(ROWS * HIDDEN + 255) / 256, 256, 0, stream>>>(x, xb, ROWS * HIDDEN);
    k_transpose_bf16<<<(HIDDEN * NQKV + 255) / 256, 256, 0, stream>>>(Wqkv, wqkvT, HIDDEN, NQKV);
    k_transpose_bf16<<<(HIDDEN * HIDDEN + 255) / 256, 256, 0, stream>>>(Wproj, wprojT, HIDDEN, HIDDEN);

    // stage 1: QKV projection (WMMA bf16, async-LDS double-buffered weights)
    k_gemm_qkv<<<(ROWS / 128) * (NQKV / 64), 256, 0, stream>>>(xb, wqkvT, bqkv, qh, kh, vtb);

    // stage 2: RoPE
    k_rope<<<(BATCH * HEADS * SEQ * 32) / 256, 256, 0, stream>>>(qh, kh);

    // stage 3: flash attention (WMMA bf16, online softmax)
    k_attn<<<(BATCH * HEADS * (SEQ / 16)) / 8, 256, 0, stream>>>(qh, kh, vtb, attn);

    // stage 4: output projection (WMMA bf16 -> f32, async-LDS weights)
    k_gemm_proj<<<(ROWS / 128) * (HIDDEN / 64), 256, 0, stream>>>(attn, wprojT, bproj, pad, out);
}
